// GatedDeltaNet_78700980732244
// MI455X (gfx1250) — matrix operand order repaired and checked by reading, not compiled, not run on hardware
//
#include <hip/hip_runtime.h>
#include <hip/hip_bf16.h>

// ---------------- types for WMMA ----------------
typedef __attribute__((ext_vector_type(16))) __bf16         v16bf;
typedef __attribute__((ext_vector_type(8)))  float          v8f;
typedef __attribute__((ext_vector_type(8)))  unsigned short v8u;
typedef __attribute__((ext_vector_type(16))) unsigned short v16u;

// ---------------- problem dims ----------------
#define HIDN   2048
#define HK     16
#define DK     128
#define NVH    32
#define DVV    128
#define KCONV  4
#define QKVW   8192                           // 2*HK*DK + NV*DV
#define NPROJ  (QKVW + NVH*DVV + NVH + NVH)   // 12352
#define BB     2
#define TTT    2048
#define MROWS  (BB*TTT)                       // 4096

__device__ __forceinline__ unsigned short f2bf(float f) {
    union { float f; unsigned int u; } x; x.f = f;
    unsigned int u = x.u;
    unsigned int r = u + 0x7FFFu + ((u >> 16) & 1u);   // round-to-nearest-even
    return (unsigned short)(r >> 16);
}

// ---------------- f32 -> bf16 conversion ----------------
__global__ void f32_to_bf16_kernel(const float* __restrict__ src,
                                   unsigned short* __restrict__ dst, size_t n) {
    size_t i = (size_t)blockIdx.x * blockDim.x + threadIdx.x;
    if (i < n) dst[i] = f2bf(src[i]);
}

// ---------------- fragment loaders ----------------
__device__ __forceinline__ v16bf load_a_frag(const unsigned short* __restrict__ p) {
    // A 16x32 bf16 layout: lanes 0-15 = rows (K 0..15),
    // lanes 16-31 = same rows (K 16..31); p is lane-resolved base.
    v8u  lo = *(const v8u*)(p);
    v8u  hi = *(const v8u*)(p + 8);
    v16u u  = __builtin_shufflevector(lo, hi,
                0,1,2,3,4,5,6,7,8,9,10,11,12,13,14,15);
    return __builtin_bit_cast(v16bf, u);
}
__device__ __forceinline__ v16bf load_b_frag(const unsigned short* __restrict__ p) {
    // B 32x16 bf16 layout: lane&15 = output column, lane>=16 holds K 16..31;
    // row-major (N,K) weights give one contiguous 32B chunk per lane.
    v16u u = *(const v16u*)(p);
    return __builtin_bit_cast(v16bf, u);
}

// ---------------- bf16 WMMA GEMM:  C(MxN) = A(MxK) * B(NxK)^T ----------------
// Register-blocked: each wave computes a 32x32 macro-tile (2 A frags x 2 B
// frags -> 4 WMMAs per 32-deep K step). Block = 8 waves as 2(M) x 4(N)
// -> 64 x 128 per workgroup. Requires M % 64 == 0, N % 32 == 0.
__global__ __launch_bounds__(256) void gemm_bf16_wmma(
    const unsigned short* __restrict__ A, const unsigned short* __restrict__ B,
    float* __restrict__ C, int M, int N, int K, int lda, int ldb, int ldc)
{
    const int lane  = threadIdx.x & 31;
    const int wave  = threadIdx.x >> 5;
    const int waveM = wave >> 2;      // 0..1
    const int waveN = wave & 3;       // 0..3
    const int m0 = blockIdx.y * 64 + waveM * 32;
    const int n0 = blockIdx.x * 128 + waveN * 32;
    if (m0 >= M || n0 >= N) return;

    const int arow  = m0 + (lane & 15);
    const int acol0 = (lane >> 4) * 16;
    const int brow  = n0 + (lane & 15);
    const int bcol0 = (lane >> 4) * 16;

    const unsigned short* ap0 = A + (size_t)arow * lda + acol0;
    const unsigned short* ap1 = ap0 + (size_t)16 * lda;
    const unsigned short* bp0 = B + (size_t)brow * ldb + bcol0;
    const unsigned short* bp1 = bp0 + (size_t)16 * ldb;

    v8f acc00 = {0.f,0.f,0.f,0.f,0.f,0.f,0.f,0.f};
    v8f acc01 = acc00, acc10 = acc00, acc11 = acc00;

    for (int k = 0; k < K; k += 32) {
        __builtin_prefetch(ap0 + 64, 0, 1);
        __builtin_prefetch(ap1 + 64, 0, 1);
        __builtin_prefetch(bp0 + 64, 0, 1);
        __builtin_prefetch(bp1 + 64, 0, 1);
        v16bf a0 = load_a_frag(ap0);
        v16bf a1 = load_a_frag(ap1);
        v16bf b0 = load_b_frag(bp0);
        v16bf b1 = load_b_frag(bp1);
        acc00 = __builtin_amdgcn_wmma_f32_16x16x32_bf16(false, a0, false, b0, (short)0, acc00, false, false);
        acc01 = __builtin_amdgcn_wmma_f32_16x16x32_bf16(false, a0, false, b1, (short)0, acc01, false, false);
        acc10 = __builtin_amdgcn_wmma_f32_16x16x32_bf16(false, a1, false, b0, (short)0, acc10, false, false);
        acc11 = __builtin_amdgcn_wmma_f32_16x16x32_bf16(false, a1, false, b1, (short)0, acc11, false, false);
        ap0 += 32; ap1 += 32; bp0 += 32; bp1 += 32;
    }

    // D layout: VGPR r holds M = (lane>>4)*8 + r, column = lane&15.
    const int rbase = (lane >> 4) * 8;
    const int cbase = lane & 15;
    #pragma unroll
    for (int mi = 0; mi < 2; ++mi) {
        v8f dlo = mi ? acc10 : acc00;
        v8f dhi = mi ? acc11 : acc01;
        float* cp = C + (size_t)(m0 + mi * 16 + rbase) * ldc + n0 + cbase;
        #pragma unroll
        for (int r = 0; r < 8; ++r) {
            cp[(size_t)r * ldc]      = dlo[r];
            cp[(size_t)r * ldc + 16] = dhi[r];
        }
    }
}

// ---------------- causal depthwise conv over time (+ new_conv_buf output) ----------------
__global__ void conv_kernel(const float* __restrict__ proj,
                            const float* __restrict__ conv_w,
                            float* __restrict__ qkvc,
                            float* __restrict__ convbuf_out)
{
    size_t idx = (size_t)blockIdx.x * blockDim.x + threadIdx.x;
    if (idx >= (size_t)BB * TTT * QKVW) return;
    int c = (int)(idx % QKVW);
    int t = (int)((idx / QKVW) % TTT);
    int b = (int)(idx / ((size_t)QKVW * TTT));
    float acc = 0.f;
    #pragma unroll
    for (int k = 0; k < KCONV; ++k) {
        int ts = t - (KCONV - 1) + k;
        if (ts >= 0)
            acc += proj[((size_t)(b * TTT + ts)) * NPROJ + c] * conv_w[c * KCONV + k];
    }
    qkvc[((size_t)(b * TTT + t)) * QKVW + c] = acc;
    if (t >= TTT - (KCONV - 1)) {
        convbuf_out[((size_t)b * QKVW + c) * (KCONV - 1) + (t - (TTT - (KCONV - 1)))] =
            proj[((size_t)(b * TTT + t)) * NPROJ + c];
    }
}

// ---------------- L2-normalize k rows in place ----------------
__global__ __launch_bounds__(128) void knorm_kernel(float* __restrict__ qkvc)
{
    __shared__ float red[128];
    const int tid = threadIdx.x;
    const int r = blockIdx.x;            // row*HK + h
    const int h = r % HK;
    const size_t row = (size_t)(r / HK);
    float* kp = qkvc + row * QKVW + HK * DK + h * DK;
    float kv = kp[tid];
    red[tid] = kv * kv;
    __syncthreads();
    for (int s = 64; s > 0; s >>= 1) {
        if (tid < s) red[tid] += red[tid + s];
        __syncthreads();
    }
    float inv = 1.f / fmaxf(sqrtf(red[0]), 1e-12f);
    kp[tid] = kv * inv;
}

// ---------------- beta = sigmoid, decay = exp(-softplus(a+dt_bias)*exp(A_log)) ----------------
__global__ void gate_kernel(const float* __restrict__ proj,
                            const float* __restrict__ dt_bias,
                            const float* __restrict__ A_log,
                            float* __restrict__ beta,
                            float* __restrict__ decay)
{
    int idx = blockIdx.x * blockDim.x + threadIdx.x;
    if (idx >= MROWS * NVH) return;
    int row = idx / NVH, n = idx % NVH;
    const float* pr = proj + (size_t)row * NPROJ;
    float bp = pr[QKVW + NVH * DVV + n];
    float ap = pr[QKVW + NVH * DVV + NVH + n] + dt_bias[n];
    beta[idx] = 1.f / (1.f + expf(-bp));
    float dt = (ap > 20.f) ? ap : log1pf(expf(ap));
    decay[idx] = expf(-dt * expf(A_log[n]));
}

// ---------------- sequential delta-rule scan ----------------
// One workgroup per (b, h). Each of 256 threads exclusively owns one state
// column S[:, c] (c = tid): the 128-float column lives in VGPRs with
// fully-unrolled static-index loops. The per-step k/q broadcast values are
// double-buffered in LDS; the NEXT step's k/q words are fetched with
// GLOBAL_LOAD_ASYNC_TO_LDS_B32 (ASYNCcnt) so global->LDS latency overlaps
// the current step's ~384 VALU ops on the serial chain.
__global__ __launch_bounds__(256) void scan_kernel(
    const float* __restrict__ qkvc, const float* __restrict__ beta,
    const float* __restrict__ decay, float* __restrict__ attn,
    float* __restrict__ state_out)
{
    __shared__ float kq[2][256];   // [buf][0..127 = k, 128..255 = q]
    const int tid = threadIdx.x;
    const int bh = blockIdx.x;
    const int b = bh >> 4;
    const int h = bh & 15;
    const int p = tid >> 7;
    const int vh = h * 2 + p;

    // this thread's k/q source column within a qkvc row
    const int srccol = (tid < 128) ? (HK * DK + h * DK + tid)
                                   : (h * DK + (tid - 128));
    // wave-relative LDS byte offsets of the two buffer slots for this thread
    const unsigned lds_slot0 = (unsigned)(uintptr_t)(&kq[0][tid]);
    const unsigned lds_slot1 = (unsigned)(uintptr_t)(&kq[1][tid]);

    float s[128];
    #pragma unroll
    for (int i = 0; i < 128; ++i) s[i] = 0.f;

    // synchronous preload of step 0
    kq[0][tid] = qkvc[(size_t)(b * TTT) * QKVW + srccol];
    __syncthreads();

    for (int t = 0; t < TTT; ++t) {
        const int cur = t & 1;
        const size_t row = (size_t)(b * TTT + t);

        // async-prefetch next step's k/q word straight into the other buffer
        if (t + 1 < TTT) {
            const float* gp = qkvc + (row + 1) * QKVW + srccol;
            unsigned lds_dst = cur ? lds_slot0 : lds_slot1;
            unsigned long long ga = (unsigned long long)(uintptr_t)gp;
            asm volatile("global_load_async_to_lds_b32 %0, %1, off"
                         :: "v"(lds_dst), "v"(ga) : "memory");
        }

        const float* ksh = kq[cur];
        const float* qsh = kq[cur] + 128;
        const float v   = qkvc[row * QKVW + 2 * HK * DK + h * 256 + tid];
        const float bet = beta[row * NVH + vh];
        const float dec = decay[row * NVH + vh];

        float sk = 0.f;
        #pragma unroll
        for (int i = 0; i < 128; ++i) sk = fmaf(ksh[i], s[i], sk);
        const float delta = (v - sk) * bet;

        float o = 0.f;
        #pragma unroll
        for (int i = 0; i < 128; ++i) {
            s[i] = fmaf(s[i], dec, ksh[i] * delta);
            o    = fmaf(qsh[i], s[i], o);
        }
        attn[row * (NVH * DVV) + h * 256 + tid] = o;

        // publish the async-filled buffer: my loads done, then all threads'
        asm volatile("s_wait_asynccnt 0x0" ::: "memory");
        __syncthreads();
    }

    #pragma unroll
    for (int i = 0; i < 128; ++i)
        state_out[((size_t)(b * HK + h) * DK + i) * 256 + tid] = s[i];
}

// ---------------- rmsnorm(out)*norm_w * silu(z) -> bf16 ----------------
__global__ __launch_bounds__(128) void norm_gate_kernel(
    const float* __restrict__ attn, const float* __restrict__ proj,
    const float* __restrict__ norm_w, unsigned short* __restrict__ gated_bf)
{
    __shared__ float red[128];
    const int tid = threadIdx.x;
    const int r = blockIdx.x;           // row*NV + vh
    const int vh = r % NVH;
    const size_t row = (size_t)(r / NVH);
    const float x = attn[row * (NVH * DVV) + vh * DVV + tid];
    red[tid] = x * x;
    __syncthreads();
    for (int s = 64; s > 0; s >>= 1) {
        if (tid < s) red[tid] += red[tid + s];
        __syncthreads();
    }
    const float rms = rsqrtf(red[0] * (1.f / 128.f) + 1e-6f);
    const float z = proj[row * NPROJ + QKVW + vh * DVV + tid];
    const float sil = z / (1.f + expf(-z));
    gated_bf[row * (NVH * DVV) + vh * DVV + tid] = f2bf(x * rms * norm_w[tid] * sil);
}

// ---------------- host-side orchestration ----------------
extern "C" void kernel_launch(void* const* d_in, const int* in_sizes, int n_in,
                              void* d_out, int out_size, void* d_ws, size_t ws_size,
                              hipStream_t stream) {
    const float* x       = (const float*)d_in[0];
    const float* W_qkv   = (const float*)d_in[1];
    const float* W_z     = (const float*)d_in[2];
    const float* W_b     = (const float*)d_in[3];
    const float* W_a     = (const float*)d_in[4];
    const float* conv_w  = (const float*)d_in[5];
    const float* dt_bias = (const float*)d_in[6];
    const float* A_log   = (const float*)d_in[7];
    const float* norm_w  = (const float*)d_in[8];
    const float* W_out   = (const float*)d_in[9];
    float* out = (float*)d_out;

    const size_t Y_SIZE     = (size_t)BB * TTT * HIDN;             // 8388608
    const size_t STATE_SIZE = (size_t)BB * HK * DK * 2 * DVV;      // 1048576

    char* ws = (char*)d_ws;
    size_t off = 0;
    auto alloc = [&](size_t bytes) {
        size_t cur = off; off += (bytes + 255) & ~(size_t)255; return cur;
    };
    unsigned short* x_bf    = (unsigned short*)(ws + alloc((size_t)MROWS * HIDN * 2));
    unsigned short* Wall_bf = (unsigned short*)(ws + alloc((size_t)NPROJ * HIDN * 2));
    unsigned short* Wout_bf = (unsigned short*)(ws + alloc((size_t)HIDN * NVH * DVV * 2));
    float* proj  = (float*)(ws + alloc((size_t)MROWS * NPROJ * 4));
    float* qkvc  = (float*)(ws + alloc((size_t)MROWS * QKVW * 4));
    float* beta  = (float*)(ws + alloc((size_t)MROWS * NVH * 4));
    float* decay = (float*)(ws + alloc((size_t)MROWS * NVH * 4));
    float* attn  = (float*)(ws + alloc((size_t)MROWS * NVH * DVV * 4));
    unsigned short* gated_bf = (unsigned short*)(ws + alloc((size_t)MROWS * NVH * DVV * 2));

    auto cvt = [&](const float* s, unsigned short* d, size_t n) {
        f32_to_bf16_kernel<<<dim3((unsigned)((n + 255) / 256)), dim3(256), 0, stream>>>(s, d, n);
    };
    cvt(x,     x_bf,                                         (size_t)MROWS * HIDN);
    cvt(W_qkv, Wall_bf,                                      (size_t)QKVW * HIDN);
    cvt(W_z,   Wall_bf + (size_t)QKVW * HIDN,                (size_t)NVH * DVV * HIDN);
    cvt(W_b,   Wall_bf + (size_t)(QKVW + NVH * DVV) * HIDN,  (size_t)NVH * HIDN);
    cvt(W_a,   Wall_bf + (size_t)(QKVW + NVH * DVV + NVH) * HIDN, (size_t)NVH * HIDN);
    cvt(W_out, Wout_bf,                                      (size_t)HIDN * NVH * DVV);

    {   // fused projection GEMM: [qkv | z | beta_pre | dt_pre]
        dim3 grid((NPROJ + 127) / 128, MROWS / 64);
        gemm_bf16_wmma<<<grid, 256, 0, stream>>>(x_bf, Wall_bf, proj,
                                                 MROWS, NPROJ, HIDN,
                                                 HIDN, HIDN, NPROJ);
    }
    {   // depthwise causal conv + new_conv_buf output
        size_t n = (size_t)BB * TTT * QKVW;
        conv_kernel<<<(unsigned)((n + 255) / 256), 256, 0, stream>>>(
            proj, conv_w, qkvc, out + Y_SIZE + STATE_SIZE);
    }
    knorm_kernel<<<MROWS * HK, 128, 0, stream>>>(qkvc);
    gate_kernel<<<(MROWS * NVH + 255) / 256, 256, 0, stream>>>(proj, dt_bias, A_log, beta, decay);

    {   // sequential scan: 32 workgroups (b,h), state in VGPRs,
        // async double-buffered k/q staging in LDS
        scan_kernel<<<BB * HK, 256, 0, stream>>>(qkvc, beta, decay, attn, out + Y_SIZE);
    }
    norm_gate_kernel<<<MROWS * NVH, 128, 0, stream>>>(attn, proj, norm_w, gated_bf);

    {   // output GEMM: y = gated @ W_out^T
        dim3 grid((HIDN + 127) / 128, MROWS / 64);
        gemm_bf16_wmma<<<grid, 256, 0, stream>>>(gated_bf, Wout_bf, out,
                                                 MROWS, HIDN, NVH * DVV,
                                                 NVH * DVV, NVH * DVV, HIDN);
    }
}
